// GraphAttentionLayer_12232066859438
// MI455X (gfx1250) — matrix operand order repaired and checked by reading, not compiled, not run on hardware
//
#include <hip/hip_runtime.h>
#include <hip/hip_bf16.h>

#define N_NODES 4096
#define F_IN    512
#define F_OUT   64
#define GAT_ALPHA 0.2f
#define NEG_BIG (-9000000000000000.0f)
#define JSPLIT  8
#define ADJ_STRIDE 36   // ints; 144B row pitch: 16B-aligned for async b128, bank-friendly

typedef __attribute__((ext_vector_type(16))) __bf16 v16bf;
typedef __attribute__((ext_vector_type(8)))  float  v8f;
typedef __attribute__((ext_vector_type(4)))  int    v4i;

#if defined(__has_builtin)
#if __has_builtin(__builtin_amdgcn_global_load_async_to_lds_b128)
#define HAVE_ASYNC_LDS 1
#endif
#endif

#define AS1V4(p) ((__attribute__((address_space(1))) v4i*)(p))
#define AS3V4(p) ((__attribute__((address_space(3))) v4i*)(p))

__device__ __forceinline__ void wait_asynccnt0() {
#if defined(__has_builtin) && __has_builtin(__builtin_amdgcn_s_wait_asynccnt)
  __builtin_amdgcn_s_wait_asynccnt(0);
#else
  asm volatile("s_wait_asynccnt 0x0" ::: "memory");
#endif
}

__device__ __forceinline__ unsigned short f2bf(float f) {
  union { float f; unsigned u; } c; c.f = f;
  unsigned r = c.u + 0x7FFFu + ((c.u >> 16) & 1u);   // round-to-nearest-even
  return (unsigned short)(r >> 16);
}

// ---------------- Stage 1: h = x @ W  (fp32), fused e1 = h @ a1 ----------------
__global__ void k_h_e1(const float* __restrict__ x, const float* __restrict__ W,
                       const float* __restrict__ a, float* __restrict__ h,
                       float* __restrict__ e1) {
  __shared__ float xs[4][F_IN];
  __shared__ float red[4][64];
  int row_l = threadIdx.x >> 6;     // 0..3
  int col   = threadIdx.x & 63;
  int row   = blockIdx.x * 4 + row_l;
  #pragma unroll
  for (int q = 0; q < 8; ++q) {
    int li = q * 256 + threadIdx.x;                 // 0..2047
    xs[li >> 9][li & 511] = x[(size_t)(blockIdx.x * 4 + (li >> 9)) * F_IN + (li & 511)];
  }
  __syncthreads();
  float acc = 0.f;
  #pragma unroll 8
  for (int k = 0; k < F_IN; ++k)
    acc = fmaf(xs[row_l][k], W[k * F_OUT + col], acc);
  h[(size_t)row * F_OUT + col] = acc;
  red[row_l][col] = acc * a[col];                   // a1 = a[0:64]
  __syncthreads();
  for (int s = 32; s > 0; s >>= 1) {
    if (col < s) red[row_l][col] += red[row_l][col + s];
    __syncthreads();
  }
  if (col == 0) e1[row] = red[row_l][0];
}

// ---------------- Stage 2a: Tb[b,t] = sum_r a2[r] * h[r*64+b, t] ----------------
__global__ void k_tb(const float* __restrict__ h, const float* __restrict__ a,
                     float* __restrict__ Tb) {
  int tid = blockIdx.x * blockDim.x + threadIdx.x;  // 4096
  int b = tid >> 6, t = tid & 63;
  float s = 0.f;
  #pragma unroll 8
  for (int r = 0; r < F_OUT; ++r)
    s = fmaf(a[F_OUT + r], h[(size_t)(r * 64 + b) * F_OUT + t], s);
  Tb[b * 64 + t] = s;
}

// ------- Stage 2b: repack h into WMMA B-fragment order (bf16, 32x16 tiles) -------
// layout: hB[((jg*4 + c)*32 + lane)*8 + v] = pack(h[jg*32+k0][n], h[jg*32+k0+1][n])
//   k0 = 2v + (lane<16 ? 0 : 16),  n = c*16 + lane%16   (contiguous k order, same as A)
__global__ void k_pack(const float* __restrict__ h, unsigned* __restrict__ hB) {
  int tid = blockIdx.x * blockDim.x + threadIdx.x;  // 131072
  int vv = tid & 7, L = (tid >> 3) & 31, c = (tid >> 8) & 3, jg = tid >> 10;
  int k0 = 2 * vv + ((L < 16) ? 0 : 16);
  int n  = c * 16 + (L & 15);
  int r0 = jg * 32 + k0;
  unsigned lo = f2bf(h[(size_t)r0 * F_OUT + n]);
  unsigned hi = f2bf(h[(size_t)(r0 + 1) * F_OUT + n]);
  hB[tid] = lo | (hi << 16);
}

// ---------------- Stage 3: per-row softmax stats (max, sum of exp) ----------------
__global__ void k_stats(const int* __restrict__ adj, const float* __restrict__ e1,
                        const float* __restrict__ Tb, float* __restrict__ rmax,
                        float* __restrict__ rsum) {
  __shared__ float tbs[64];
  __shared__ float red[256];
  int i = blockIdx.x;
  int t = threadIdx.x;
  if (t < 64) tbs[t] = Tb[(i >> 6) * 64 + t];
  __syncthreads();
  float ev[16];
  float m = NEG_BIG;
  #pragma unroll
  for (int s = 0; s < 16; ++s) {
    int j = s * 256 + t;
    float e = e1[j] + tbs[j & 63];
    e = e > 0.f ? e : GAT_ALPHA * e;
    e = (adj[(size_t)i * N_NODES + j] > 0) ? e : NEG_BIG;
    ev[s] = e;
    m = fmaxf(m, e);
  }
  red[t] = m;
  __syncthreads();
  for (int s = 128; s > 0; s >>= 1) { if (t < s) red[t] = fmaxf(red[t], red[t + s]); __syncthreads(); }
  m = red[0];
  __syncthreads();
  float sum = 0.f;
  #pragma unroll
  for (int s = 0; s < 16; ++s) sum += __expf(ev[s] - m);
  red[t] = sum;
  __syncthreads();
  for (int s = 128; s > 0; s >>= 1) { if (t < s) red[t] += red[t + s]; __syncthreads(); }
  if (t == 0) { rmax[i] = m; rsum[i] = red[0]; }
}

// ------- Stage 4: fused attention-weighted accumulation, bf16 WMMA 16x16x32 -------
__global__ void __launch_bounds__(32)
k_attn(const int* __restrict__ adj, const float* __restrict__ e1,
       const float* __restrict__ Tb, const float* __restrict__ rmax,
       const unsigned* __restrict__ hB, float* __restrict__ accS) {
  __shared__ int adjS[16 * ADJ_STRIDE];             // 144B pitch: async-b128 aligned
  int L = threadIdx.x;
  int i0 = blockIdx.x * 16;                         // 16-row tile
  int tbrow = (i0 >> 6) * 64;
  float m = rmax[i0 + (L & 15)];                    // this lane only ever touches row L%16 of A
  int hi8 = (L < 16) ? 0 : 8;
  int M = L & 15;
  v8f acc[4] = {};                                  // C/D: 4 chunks of 16 cols, f32 accum
  const uint4* hB4 = (const uint4*)hB;

  for (int it = 0; it < 16; ++it) {
    int jg = blockIdx.y * 16 + it;
    int J = jg * 32;                                // K-step of 32
    // stage 16x32 adj tile into LDS (async direct-to-LDS when available)
    #pragma unroll
    for (int q = 0; q < 4; ++q) {
      int li = q * 32 + L;                          // int4 index 0..127
      int r = li >> 3;
      int c4 = (li & 7) * 4;
      const int* gsrc = adj + (size_t)(i0 + r) * N_NODES + J + c4;
#ifdef HAVE_ASYNC_LDS
      __builtin_amdgcn_global_load_async_to_lds_b128(
          AS1V4(gsrc), AS3V4(&adjS[r * ADJ_STRIDE + c4]), 0, 0);
#else
      int4 vdat = *(const int4*)gsrc;
      adjS[r * ADJ_STRIDE + c4 + 0] = vdat.x;
      adjS[r * ADJ_STRIDE + c4 + 1] = vdat.y;
      adjS[r * ADJ_STRIDE + c4 + 2] = vdat.z;
      adjS[r * ADJ_STRIDE + c4 + 3] = vdat.w;
#endif
      if (it < 15)                                  // prefetch next tile -> global_prefetch_b8
        __builtin_prefetch(gsrc + 32, 0, 0);
    }
#ifdef HAVE_ASYNC_LDS
    wait_asynccnt0();
#endif
    __syncthreads();
    // build A fragment: P = exp(masked lrelu(e1[j] + Tb[...]) - rowmax), as bf16
    v16bf Avec;
    #pragma unroll
    for (int vv = 0; vv < 8; ++vv) {
      int k0 = 2 * vv + 2 * hi8;
      #pragma unroll
      for (int p = 0; p < 2; ++p) {
        int k = k0 + p;
        int j = J + k;
        float e = e1[j] + Tb[tbrow + (j & 63)];
        e = e > 0.f ? e : GAT_ALPHA * e;
        e = (adjS[M * ADJ_STRIDE + k] > 0) ? e : NEG_BIG;
        Avec[2 * vv + p] = (__bf16)__expf(e - m);   // native v_cvt -> bf16; masked -> 0
      }
    }
    __syncthreads();                                // drain DScnt before next async writes
    // 4 B fragments (pre-swizzled, coalesced) + 4 WMMAs
    #pragma unroll
    for (int c = 0; c < 4; ++c) {
      union { uint4 q[2]; v16bf vec; } B;
      size_t u4 = ((size_t)(jg * 4 + c) * 32 + L) * 2;
      B.q[0] = hB4[u4];
      B.q[1] = hB4[u4 + 1];
      acc[c] = __builtin_amdgcn_wmma_f32_16x16x32_bf16(
                 false, Avec, false, B.vec, (short)0, acc[c], false, false);
    }
  }
  // write deterministic per-split partials (no float atomics -> replay-stable)
  float* dst = accS + (size_t)blockIdx.y * (N_NODES * F_OUT);
  #pragma unroll
  for (int c = 0; c < 4; ++c) {
    #pragma unroll
    for (int d = 0; d < 8; ++d) {
      int row = i0 + d + hi8;                       // C/D layout: VGPR d -> M=d / d+8
      int col = c * 16 + (L & 15);
      dst[(size_t)row * F_OUT + col] = acc[c][d];
    }
  }
}

// ---------------- Stage 5: reduce splits, /rowsum, relu, +bias ----------------
__global__ void k_final(const float* __restrict__ accS, const float* __restrict__ rsum,
                        const float* __restrict__ bias, float* __restrict__ out) {
  int tid = blockIdx.x * 256 + threadIdx.x;         // 262144
  float s = 0.f;
  #pragma unroll
  for (int sp = 0; sp < JSPLIT; ++sp) s += accS[(size_t)sp * (N_NODES * F_OUT) + tid];
  int i = tid >> 6, t = tid & 63;
  float v = s / rsum[i];
  out[tid] = fmaxf(v, 0.f) + bias[t];
}

extern "C" void kernel_launch(void* const* d_in, const int* in_sizes, int n_in,
                              void* d_out, int out_size, void* d_ws, size_t ws_size,
                              hipStream_t stream) {
  const float* x    = (const float*)d_in[0];
  const int*   adj  = (const int*)  d_in[1];
  const float* W    = (const float*)d_in[2];
  const float* a    = (const float*)d_in[3];
  const float* bias = (const float*)d_in[4];
  float* out = (float*)d_out;

  float*    h    = (float*)d_ws;                    // 262144 f
  unsigned* hB   = (unsigned*)(h + 262144);         // 131072 u32 (bf16 pairs)
  float*    e1   = (float*)(hB + 131072);           // 4096 f
  float*    Tb   = e1 + 4096;                       // 4096 f
  float*    rmax = Tb + 4096;                       // 4096 f
  float*    rsum = rmax + 4096;                     // 4096 f
  float*    accS = rsum + 4096;                     // JSPLIT * 262144 f

  k_h_e1 <<<1024, 256, 0, stream>>>(x, W, a, h, e1);
  k_tb   <<<16,   256, 0, stream>>>(h, a, Tb);
  k_pack <<<512,  256, 0, stream>>>(h, hB);
  k_stats<<<4096, 256, 0, stream>>>(adj, e1, Tb, rmax, rsum);
  k_attn <<<dim3(256, JSPLIT), 32, 0, stream>>>(adj, e1, Tb, rmax, hB, accS);
  k_final<<<1024, 256, 0, stream>>>(accS, rsum, bias, out);
}
